// DenseTrajectory_47107201302857
// MI455X (gfx1250) — matrix operand order repaired and checked by reading, not compiled, not run on hardware
//
#include <hip/hip_runtime.h>

#define BB 2
#define CC 32
#define TT 8
#define HH 192
#define WW 352
#define HW (HH*WW)
#define PIX (BB*HH*WW)

typedef __attribute__((ext_vector_type(16))) __bf16 v16bf;
typedef __attribute__((ext_vector_type(8)))  __bf16 v8bf;
typedef __attribute__((ext_vector_type(8)))  float  v8f;

// ---------------------------------------------------------------------------
// Bilinear sampling setup (grid_sample, padding_mode='zeros', align_corners=False)
// ---------------------------------------------------------------------------
struct Bilin {
    int   i00, i01, i10, i11;
    float w00, w01, w10, w11;
};

__device__ inline Bilin make_bilin(float xs, float ys) {
    Bilin s;
    float x0f = floorf(xs), y0f = floorf(ys);
    float fx = xs - x0f,    fy = ys - y0f;
    float x1f = x0f + 1.f,  y1f = y0f + 1.f;
    bool vx0 = (x0f >= 0.f) && (x0f <= (float)(WW - 1));
    bool vx1 = (x1f >= 0.f) && (x1f <= (float)(WW - 1));
    bool vy0 = (y0f >= 0.f) && (y0f <= (float)(HH - 1));
    bool vy1 = (y1f >= 0.f) && (y1f <= (float)(HH - 1));
    int x0 = min(max((int)x0f, 0), WW - 1);
    int x1 = min(max((int)x1f, 0), WW - 1);
    int y0 = min(max((int)y0f, 0), HH - 1);
    int y1 = min(max((int)y1f, 0), HH - 1);
    s.i00 = y0 * WW + x0;  s.i01 = y0 * WW + x1;
    s.i10 = y1 * WW + x0;  s.i11 = y1 * WW + x1;
    s.w00 = (1.f - fx) * (1.f - fy) * ((vx0 && vy0) ? 1.f : 0.f);
    s.w01 = fx * (1.f - fy)         * ((vx1 && vy0) ? 1.f : 0.f);
    s.w10 = (1.f - fx) * fy         * ((vx0 && vy1) ? 1.f : 0.f);
    s.w11 = fx * fy                 * ((vx1 && vy1) ? 1.f : 0.f);
    return s;
}

__device__ inline float bsamp(const float* __restrict__ p, const Bilin& s) {
    return p[s.i00]*s.w00 + p[s.i01]*s.w01 + p[s.i10]*s.w10 + p[s.i11]*s.w11;
}

// vgrid -> grid_sample pixel coords (two-step, matching the reference literally)
__device__ inline void flow_to_xy(float px, float py, float fx_, float fy_,
                                  float& xs, float& ys) {
    float vx = px + fx_, vy = py + fy_;
    float gx = 2.f * vx / (float)(WW - 1) - 1.f;
    float gy = 2.f * vy / (float)(HH - 1) - 1.f;
    xs = ((gx + 1.f) * (float)WW - 1.f) * 0.5f;
    ys = ((gy + 1.f) * (float)HH - 1.f) * 0.5f;
}

// ---------------------------------------------------------------------------
// Weight reorder: W_rnn [32][64][3][3] (OIHW) -> Wb [o=32][k=576] bf16,
// k = (ky*3+kx)*64 + cin  (tap-major, channel contiguous)
// ---------------------------------------------------------------------------
__global__ void prep_weights_k(const float* __restrict__ Wrnn, __bf16* __restrict__ Wb) {
    int i = blockIdx.x * blockDim.x + threadIdx.x;
    if (i >= 32 * 576) return;
    int o = i / 576, k = i % 576;
    int tap = k >> 6, cin = k & 63;
    int ky = tap / 3, kx = tap % 3;
    Wb[i] = (__bf16)Wrnn[((o * 64 + cin) * 3 + ky) * 3 + kx];
}

// ---------------------------------------------------------------------------
// t = 0 init: zin = [zeros(32) | feat_t0] (bf16, channel-last), masks[0] = fg0>0.5
// ---------------------------------------------------------------------------
__global__ void init_t0_k(const float* __restrict__ feat, const float* __restrict__ fg,
                          __bf16* __restrict__ zin, float* __restrict__ masks) {
    int i = blockIdx.x * blockDim.x + threadIdx.x;
    if (i >= PIX) return;
    int x = i % WW, y = (i / WW) % HH, b = i / HW;
    __bf16* zp = zin + (size_t)i * 64;
#pragma unroll
    for (int c = 0; c < 32; ++c) zp[c] = (__bf16)0.0f;
#pragma unroll 4
    for (int c = 0; c < 32; ++c)
        zp[32 + c] = (__bf16)feat[((((size_t)b * CC + c) * TT + 0) * HH + y) * WW + x];
    float m = fg[(((size_t)b * TT + 0) * HH + y) * WW + x] > 0.5f ? 1.f : 0.f;
    masks[(((size_t)0 * BB + b) * HH + y) * WW + x] = m;
}

// ---------------------------------------------------------------------------
// Warp step (t >= 1): mask logic + bilinear warp of h/cw, writes zin
// ---------------------------------------------------------------------------
__global__ void warp_step_k(const float* __restrict__ feat, const float* __restrict__ fg,
                            const float* __restrict__ flow, const float* __restrict__ invf,
                            const float* __restrict__ hprev, const float* __restrict__ cprev,
                            float* __restrict__ hcur, float* __restrict__ ccur,
                            float* __restrict__ masks, __bf16* __restrict__ zin, int t) {
    int i = blockIdx.x * blockDim.x + threadIdx.x;
    if (i >= PIX) return;
    int x = i % WW, y = (i / WW) % HH, b = i / HW;

    size_t ifb = ((((size_t)b * 2 + 0) * TT + t) * HH + y) * WW + x;
    float ivx = invf[ifb];
    float ivy = invf[ifb + (size_t)TT * HW];
    float xs, ys;
    flow_to_xy((float)x, (float)y, ivx, ivy, xs, ys);
    Bilin s = make_bilin(xs, ys);

    // warped binarized fg[t-1]
    const float* fgp = fg + (((size_t)b * TT + (t - 1)) * HW);
    float wfg = (fgp[s.i00] > 0.5f ? s.w00 : 0.f) + (fgp[s.i01] > 0.5f ? s.w01 : 0.f)
              + (fgp[s.i10] > 0.5f ? s.w10 : 0.f) + (fgp[s.i11] > 0.5f ? s.w11 : 0.f);
    float fgb = fg[(((size_t)b * TT + t) * HH + y) * WW + x] > 0.5f ? 1.f : 0.f;
    float mask = (wfg > 0.5f ? 1.f : 0.f) * fgb;

    // flow consistency: warped forward flow(t-1) vs inv flow(t)
    const float* flx = flow + ((((size_t)b * 2 + 0) * TT + (t - 1)) * HW);
    const float* fly = flow + ((((size_t)b * 2 + 1) * TT + (t - 1)) * HW);
    float wfx = bsamp(flx, s), wfy = bsamp(fly, s);
    float dx = wfx + ivx, dy = wfy + ivy;
    float diff = dx * dx + dy * dy;
    float thr  = 0.01f * (wfx * wfx + ivx * ivx + wfy * wfy + ivy * ivy) + 0.5f;
    if (diff > thr) mask = 0.f;
    masks[(((size_t)t * BB + b) * HH + y) * WW + x] = mask;

    __bf16* zp = zin + (size_t)i * 64;
    const float* hp = hprev + (size_t)b * CC * HW;
    const float* cp = cprev + (size_t)b * CC * HW;
    float* ho = hcur + (size_t)b * CC * HW + (size_t)y * WW + x;
    float* co = ccur + (size_t)b * CC * HW + (size_t)y * WW + x;
#pragma unroll 4
    for (int c = 0; c < CC; ++c) {
        float wh = bsamp(hp + (size_t)c * HW, s) * mask;
        float wc = bsamp(cp + (size_t)c * HW, s) * mask;
        ho[(size_t)c * HW] = wh;
        co[(size_t)c * HW] = wc;
        zp[c] = (__bf16)(wh / fmaxf(wc, 1e-6f));
        zp[32 + c] = (__bf16)feat[((((size_t)b * CC + c) * TT + t) * HH + y) * WW + x];
    }
}

// ---------------------------------------------------------------------------
// Conv 3x3 (64 -> 32 ch, SAME, zero pad) as WMMA implicit GEMM + fused gate.
// Block = 256 thr = 8 waves; wave w owns 16 pixels of row (tileY*8 + w).
// Per wave: two 16x16 f32 accumulators (out ch 0-15 / 16-31), K = 576 in 18
// steps of v_wmma_f32_16x16x32_bf16.
// ---------------------------------------------------------------------------
#define CR 8
#define CCOL 16

__global__ __launch_bounds__(256)
void conv_gate_k(const __bf16* __restrict__ zin,   // [B][H][W][64] bf16
                 const __bf16* __restrict__ wmat,  // [32][576]     bf16
                 const float* __restrict__ feat,   // [B][C][T][H][W]
                 float* __restrict__ hbuf,         // [B][32][H][W] (holds wh on entry)
                 float* __restrict__ cbuf,         // [B][32][H][W] (holds wc on entry)
                 int t, int is_first) {
    __shared__ __align__(16) __bf16 zs[(CR + 2) * (CCOL + 2) * 64];  // 23040 B
    __shared__ __align__(16) __bf16 wsh[32 * 576];                   // 36864 B

    const int b  = blockIdx.z;
    const int y0 = blockIdx.y * CR;
    const int x0 = blockIdx.x * CCOL;
    const int tid = threadIdx.x;

    // stage weights (contiguous)
    {
        const uint4* src = (const uint4*)wmat;
        uint4* dst = (uint4*)wsh;
#pragma unroll 3
        for (int i = tid; i < 32 * 576 * 2 / 16; i += 256) dst[i] = src[i];
    }
    // stage z tile with zero halo; 16B (8 bf16) chunks
    {
        uint4* dst = (uint4*)zs;
        const int nchunk = (CR + 2) * (CCOL + 2) * 8;  // 1440
        for (int i = tid; i < nchunk; i += 256) {
            int ch8 = i & 7;
            int pix = i >> 3;
            int col = pix % (CCOL + 2);
            int row = pix / (CCOL + 2);
            int gy = y0 - 1 + row;
            int gx = x0 - 1 + col;
            uint4 v = make_uint4(0u, 0u, 0u, 0u);
            if (gy >= 0 && gy < HH && gx >= 0 && gx < WW)
                v = *(const uint4*)(zin + ((size_t)((b * HH + gy) * WW + gx)) * 64 + ch8 * 8);
            dst[i] = v;
        }
    }
    __syncthreads();

    const int wv   = tid >> 5;        // wave = row within tile
    const int lane = tid & 31;
    const int px   = lane & 15;       // A: M = px   |  B: N = px
    const int lh   = lane >> 4;       // lane half

    v8f acc0 = {0.f, 0.f, 0.f, 0.f, 0.f, 0.f, 0.f, 0.f};
    v8f acc1 = {0.f, 0.f, 0.f, 0.f, 0.f, 0.f, 0.f, 0.f};

#pragma unroll
    for (int kb = 0; kb < 18; ++kb) {
        const int tap = kb >> 1, half = kb & 1;
        const int ky = tap / 3, kx = tap % 3;
        // A fragment: 16-bit layout -> lane half holds K' {0..7,16..23} / {8..15,24..31}
        const __bf16* za = &zs[((wv + ky) * (CCOL + 2) + (px + kx)) * 64 + half * 32 + lh * 8];
        v8bf alo = *(const v8bf*)za;
        v8bf ahi = *(const v8bf*)(za + 16);
        v16bf a = __builtin_shufflevector(alo, ahi, 0,1,2,3,4,5,6,7,8,9,10,11,12,13,14,15);
        // B fragments: same 16-bit layout as A -> lane half holds K' {0..7,16..23} / {8..15,24..31}
        const __bf16* wb0 = &wsh[(0 * 16 + px) * 576 + kb * 32 + lh * 8];
        v8bf b0lo = *(const v8bf*)wb0;
        v8bf b0hi = *(const v8bf*)(wb0 + 16);
        v16bf bf0 = __builtin_shufflevector(b0lo, b0hi, 0,1,2,3,4,5,6,7,8,9,10,11,12,13,14,15);
        acc0 = __builtin_amdgcn_wmma_f32_16x16x32_bf16(false, a, false, bf0,
                                                       (short)0, acc0, false, false);
        const __bf16* wb1 = &wsh[(16 + px) * 576 + kb * 32 + lh * 8];
        v8bf b1lo = *(const v8bf*)wb1;
        v8bf b1hi = *(const v8bf*)(wb1 + 16);
        v16bf bf1 = __builtin_shufflevector(b1lo, b1hi, 0,1,2,3,4,5,6,7,8,9,10,11,12,13,14,15);
        acc1 = __builtin_amdgcn_wmma_f32_16x16x32_bf16(false, a, false, bf1,
                                                       (short)0, acc1, false, false);
    }

    // C/D layout: VGPR j -> pixel M = lh*8 + j, channel N = px (+16 for acc1)
    const int y = y0 + wv;
#pragma unroll
    for (int j = 0; j < 8; ++j) {
        const int x = x0 + lh * 8 + j;
        const float w0 = 1.f / (1.f + __expf(-acc0[j]));
        const float w1 = 1.f / (1.f + __expf(-acc1[j]));
        const size_t fi0 = ((((size_t)b * CC + px)        * TT + t) * HH + y) * WW + x;
        const size_t fi1 = ((((size_t)b * CC + (16 + px)) * TT + t) * HH + y) * WW + x;
        const size_t hi0 = (((size_t)b * CC + px)        * HH + y) * WW + x;
        const size_t hi1 = (((size_t)b * CC + (16 + px)) * HH + y) * WW + x;
        const float f0 = feat[fi0], f1 = feat[fi1];
        const float wh0 = is_first ? 0.f : hbuf[hi0];
        const float wc0 = is_first ? 0.f : cbuf[hi0];
        const float wh1 = is_first ? 0.f : hbuf[hi1];
        const float wc1 = is_first ? 0.f : cbuf[hi1];
        hbuf[hi0] = wh0 + w0 * f0;  cbuf[hi0] = wc0 + w0;
        hbuf[hi1] = wh1 + w1 * f1;  cbuf[hi1] = wc1 + w1;
    }
}

// ---------------------------------------------------------------------------
// Location recurrence
// ---------------------------------------------------------------------------
__global__ void loc_init_k(const float* __restrict__ flow, float* __restrict__ loc) {
    int i = blockIdx.x * blockDim.x + threadIdx.x;
    if (i >= PIX) return;
    int x = i % WW, y = (i / WW) % HH, b = i / HW;
    float gx = 2.f * (float)x / (float)(WW - 1) - 1.f;
    float gy = 2.f * (float)y / (float)(HH - 1) - 1.f;
    float flx = flow[((((size_t)b * 2 + 0) * TT + 0) * HH + y) * WW + x];
    float fly = flow[((((size_t)b * 2 + 1) * TT + 0) * HH + y) * WW + x];
    float* lp = loc + (size_t)b * 5 * HW + (size_t)y * WW + x;
    lp[0] = gx; lp[HW] = gy; lp[2 * HW] = flx; lp[3 * HW] = fly; lp[4 * HW] = 1.f;
}

__global__ void loc_step_k(const float* __restrict__ flow, const float* __restrict__ invf,
                           const float* __restrict__ masks,
                           const float* __restrict__ lprev, float* __restrict__ lcur, int t) {
    int i = blockIdx.x * blockDim.x + threadIdx.x;
    if (i >= PIX) return;
    int x = i % WW, y = (i / WW) % HH, b = i / HW;
    size_t ifb = ((((size_t)b * 2 + 0) * TT + t) * HH + y) * WW + x;
    float ivx = invf[ifb];
    float ivy = invf[ifb + (size_t)TT * HW];
    float xs, ys;
    flow_to_xy((float)x, (float)y, ivx, ivy, xs, ys);
    Bilin s = make_bilin(xs, ys);
    float m = masks[(((size_t)t * BB + b) * HH + y) * WW + x];
    const float* lp = lprev + (size_t)b * 5 * HW;
    float o[5];
#pragma unroll
    for (int c = 0; c < 5; ++c) o[c] = bsamp(lp + (size_t)c * HW, s) * m;
    float gx = 2.f * (float)x / (float)(WW - 1) - 1.f;
    float gy = 2.f * (float)y / (float)(HH - 1) - 1.f;
    o[0] += gx;
    o[1] += gy;
    o[2] += flow[((((size_t)b * 2 + 0) * TT + t) * HH + y) * WW + x];
    o[3] += flow[((((size_t)b * 2 + 1) * TT + t) * HH + y) * WW + x];
    o[4] += 1.f;
    float* lo_ = lcur + (size_t)b * 5 * HW + (size_t)y * WW + x;
#pragma unroll
    for (int c = 0; c < 5; ++c) lo_[(size_t)c * HW] = o[c];
}

// ---------------------------------------------------------------------------
// emb[t] = h/clip(cw) + W2 @ leaky_relu(W1 @ (loc[:4]/loc[4]))
// ---------------------------------------------------------------------------
__global__ __launch_bounds__(256)
void emb_k(const float* __restrict__ h, const float* __restrict__ cw,
           const float* __restrict__ loc, const float* __restrict__ W1,
           const float* __restrict__ W2, float* __restrict__ emb, int t) {
    __shared__ float sW1[CC * 4];
    __shared__ float sW2[CC * CC];
    int tid = threadIdx.x;
    for (int i = tid; i < CC * 4; i += 256) sW1[i] = W1[i];
    for (int i = tid; i < CC * CC; i += 256) sW2[i] = W2[i];
    __syncthreads();
    int i = blockIdx.x * blockDim.x + tid;
    if (i >= PIX) return;
    int x = i % WW, y = (i / WW) % HH, b = i / HW;
    const float* lp = loc + (size_t)b * 5 * HW + (size_t)y * WW + x;
    float inv = 1.f / lp[4 * HW];
    float v0 = lp[0] * inv, v1 = lp[HW] * inv, v2 = lp[2 * HW] * inv, v3 = lp[3 * HW] * inv;
    float u[CC];
#pragma unroll
    for (int j = 0; j < CC; ++j) {
        float a = sW1[j * 4 + 0] * v0 + sW1[j * 4 + 1] * v1
                + sW1[j * 4 + 2] * v2 + sW1[j * 4 + 3] * v3;
        u[j] = a > 0.f ? a : 0.01f * a;
    }
    const float* hp = h  + (size_t)b * CC * HW + (size_t)y * WW + x;
    const float* cp = cw + (size_t)b * CC * HW + (size_t)y * WW + x;
#pragma unroll 2
    for (int c = 0; c < CC; ++c) {
        float e = 0.f;
#pragma unroll
        for (int j = 0; j < CC; ++j) e += sW2[c * CC + j] * u[j];
        float hv = hp[(size_t)c * HW];
        float cv = cp[(size_t)c * HW];
        emb[((((size_t)b * CC + c) * TT + t) * HH + y) * WW + x] = hv / fmaxf(cv, 1e-12f) + e;
    }
}

// ---------------------------------------------------------------------------
// Tails: t<T-1: fg[t]>0.5 && !(warp(masks[t+1], flow[t])>0.5); t=T-1: fg>0.5
// ---------------------------------------------------------------------------
__global__ void tails_k(const float* __restrict__ fg, const float* __restrict__ flow,
                        const float* __restrict__ masks, float* __restrict__ tails) {
    int i = blockIdx.x * blockDim.x + threadIdx.x;
    if (i >= TT * PIX) return;
    int x = i % WW, y = (i / WW) % HH, t = (i / HW) % TT, b = i / (HW * TT);
    float fgv = fg[(((size_t)b * TT + t) * HH + y) * WW + x] > 0.5f ? 1.f : 0.f;
    float tail;
    if (t == TT - 1) {
        tail = fgv;
    } else {
        float flx = flow[((((size_t)b * 2 + 0) * TT + t) * HH + y) * WW + x];
        float fly = flow[((((size_t)b * 2 + 1) * TT + t) * HH + y) * WW + x];
        float xs, ys;
        flow_to_xy((float)x, (float)y, flx, fly, xs, ys);
        Bilin s = make_bilin(xs, ys);
        const float* mp = masks + (((size_t)(t + 1) * BB + b) * HW);
        float w = bsamp(mp, s) > 0.5f ? 1.f : 0.f;
        tail = (fgv - w == 1.f) ? 1.f : 0.f;
    }
    tails[(((size_t)b * TT + t) * HH + y) * WW + x] = tail;
}

// ---------------------------------------------------------------------------
extern "C" void kernel_launch(void* const* d_in, const int* in_sizes, int n_in,
                              void* d_out, int out_size, void* d_ws, size_t ws_size,
                              hipStream_t stream) {
    const float* feat = (const float*)d_in[0];
    const float* fg   = (const float*)d_in[1];
    const float* flow = (const float*)d_in[2];
    const float* invf = (const float*)d_in[3];
    const float* Wrnn = (const float*)d_in[4];
    const float* W1   = (const float*)d_in[5];
    const float* W2   = (const float*)d_in[6];

    float* out       = (float*)d_out;
    float* out_fg    = out;
    float* out_emb   = out + (size_t)BB * TT * HW;
    float* out_tails = out_emb + (size_t)BB * CC * TT * HW;

    char* wsp = (char*)d_ws;
    size_t off = 0;
    auto carve = [&](size_t bytes) -> char* {
        char* p = wsp + off;
        off += (bytes + 255) & ~(size_t)255;
        return p;
    };
    __bf16* Wb  = (__bf16*)carve((size_t)32 * 576 * 2);
    __bf16* zin = (__bf16*)carve((size_t)PIX * 64 * 2);
    float* hbuf[2] = {(float*)carve((size_t)BB * CC * HW * 4),
                      (float*)carve((size_t)BB * CC * HW * 4)};
    float* cbuf[2] = {(float*)carve((size_t)BB * CC * HW * 4),
                      (float*)carve((size_t)BB * CC * HW * 4)};
    float* masks = (float*)carve((size_t)TT * BB * HW * 4);
    float* locb[2] = {(float*)carve((size_t)BB * 5 * HW * 4),
                      (float*)carve((size_t)BB * 5 * HW * 4)};

    const int pixBlocks = (PIX + 255) / 256;
    dim3 convGrid(WW / CCOL, HH / CR, BB);

    prep_weights_k<<<(32 * 576 + 255) / 256, 256, 0, stream>>>(Wrnn, Wb);

    // t = 0
    init_t0_k<<<pixBlocks, 256, 0, stream>>>(feat, fg, zin, masks);
    conv_gate_k<<<convGrid, 256, 0, stream>>>(zin, Wb, feat, hbuf[0], cbuf[0], 0, 1);
    loc_init_k<<<pixBlocks, 256, 0, stream>>>(flow, locb[0]);
    emb_k<<<pixBlocks, 256, 0, stream>>>(hbuf[0], cbuf[0], locb[0], W1, W2, out_emb, 0);

    // t = 1..T-1 (sequential recurrence)
    for (int t = 1; t < TT; ++t) {
        int cur = t & 1, prev = cur ^ 1;
        warp_step_k<<<pixBlocks, 256, 0, stream>>>(feat, fg, flow, invf,
                                                   hbuf[prev], cbuf[prev],
                                                   hbuf[cur], cbuf[cur],
                                                   masks, zin, t);
        conv_gate_k<<<convGrid, 256, 0, stream>>>(zin, Wb, feat, hbuf[cur], cbuf[cur], t, 0);
        loc_step_k<<<pixBlocks, 256, 0, stream>>>(flow, invf, masks, locb[prev], locb[cur], t);
        emb_k<<<pixBlocks, 256, 0, stream>>>(hbuf[cur], cbuf[cur], locb[cur], W1, W2, out_emb, t);
    }

    tails_k<<<(TT * PIX + 255) / 256, 256, 0, stream>>>(fg, flow, masks, out_tails);
    hipMemcpyAsync(out_fg, fg, (size_t)BB * TT * HW * sizeof(float),
                   hipMemcpyDeviceToDevice, stream);
}
